// QuantumKernel_7851200217455
// MI455X (gfx1250) — matrix operand order repaired and checked by reading, not compiled, not run on hardware
//
#include <hip/hip_runtime.h>
#include <hip/hip_bf16.h>

// ---------------------------------------------------------------------------
// Quantum conv layer on MI455X (gfx1250).
//
// Reformulation:
//   state_after_embedding psi_p  = closed-form Kronecker product (RX on |0..0>)
//   rest of circuit              = fixed 512x512 unitary W (weights only)
//   z_p = sum_j sign(j) |(W psi_p)_j|^2 ,  sign = +1 for j<256 (wire0 MSB = 0)
//
// => one complex GEMM  Phi = W * Psi  (M=512, K=512, N=32768), done with
//    v_wmma_f32_16x16x32_f16 (f16 storage, f32 accumulate).
//
// Schedule (kernel 2): each block = 64 patches (4 tiles of 16) with Psi in
// LDS; each wave owns 4 row-tiles and sweeps all 4 patch tiles per A-fragment
// load, so one W fragment feeds 16 WMMAs (4x less L2 traffic than a
// patch-tile-per-wave split). W (1 MB f16) is L2-resident throughout.
// ---------------------------------------------------------------------------

#define NQ   9
#define DIM  512            // 2^9
#define HDIM 64             // image H = W = 64

typedef __attribute__((ext_vector_type(16))) _Float16 v16h;
typedef __attribute__((ext_vector_type(8)))  _Float16 v8h;
typedef __attribute__((ext_vector_type(8)))  float    v8f;

// ------------------------- complex helpers (f32) ---------------------------
__device__ __forceinline__ float2 cmul(float2 a, float2 b) {
    return make_float2(a.x * b.x - a.y * b.y, a.x * b.y + a.y * b.x);
}
__device__ __forceinline__ float2 cadd(float2 a, float2 b) {
    return make_float2(a.x + b.x, a.y + b.y);
}
__device__ __forceinline__ float2 cexpi(float t) {   // e^{i t}
    return make_float2(__cosf(t), __sinf(t));
}
__device__ __forceinline__ float2 cscale(float2 a, float s) {
    return make_float2(a.x * s, a.y * s);
}

// ===========================================================================
// Kernel 1: build W (512x512 complex) = StronglyEntanglingLayers(2 layers).
// Each block evolves 32 basis columns held in LDS (128 KB).
// Wire q maps to bit (8 - q) of the amplitude index (wire 0 = MSB).
// ===========================================================================

__device__ __forceinline__ void apply1q(float2 (*st)[32], int col, int grp,
                                        int bit, float2 u00, float2 u01,
                                        float2 u10, float2 u11) {
    // 256 amplitude pairs, split over the 8 thread-groups (disjoint pairs).
    for (int p = grp; p < 256; p += 8) {
        int j0 = ((p >> bit) << (bit + 1)) | (p & ((1 << bit) - 1));
        int j1 = j0 | (1 << bit);
        float2 a0 = st[j0][col], a1 = st[j1][col];
        st[j0][col] = cadd(cmul(u00, a0), cmul(u01, a1));
        st[j1][col] = cadd(cmul(u10, a0), cmul(u11, a1));
    }
}

__device__ __forceinline__ void applyCnot(float2 (*st)[32], int col, int grp,
                                          int cb, int tb) {
    // 128 swap pairs: control bit set, target bit {0,1}.
    for (int i = grp; i < 128; i += 8) {
        int j0 = 1 << cb, rem = i;
        #pragma unroll
        for (int b = 0; b < NQ; ++b) {
            if (b == cb || b == tb) continue;
            j0 |= (rem & 1) << b;
            rem >>= 1;
        }
        int j1 = j0 | (1 << tb);
        float2 t = st[j0][col];
        st[j0][col] = st[j1][col];
        st[j1][col] = t;
    }
}

__global__ __launch_bounds__(256) void build_W(const float* __restrict__ w,
                                               _Float16* __restrict__ Wr,
                                               _Float16* __restrict__ Wi) {
    __shared__ float2 st[DIM][32];                 // 128 KB
    const int tid = threadIdx.x;
    const int col = tid & 31;                      // local column
    const int grp = tid >> 5;                      // 8 row-groups
    const int colBase = blockIdx.x * 32;           // global basis column

    for (int j = grp; j < DIM; j += 8)
        st[j][col] = (j == colBase + col) ? make_float2(1.f, 0.f)
                                          : make_float2(0.f, 0.f);
    __syncthreads();

    for (int l = 0; l < 2; ++l) {
        // Rot(phi, theta, omega) = RZ(omega) RY(theta) RZ(phi) on every wire
        for (int q = 0; q < NQ; ++q) {
            const float phi = w[(l * NQ + q) * 3 + 0];
            const float th  = w[(l * NQ + q) * 3 + 1];
            const float om  = w[(l * NQ + q) * 3 + 2];
            const float c = __cosf(0.5f * th), s = __sinf(0.5f * th);
            float2 u00 = cscale(cexpi(-0.5f * (phi + om)),  c);
            float2 u01 = cscale(cexpi( 0.5f * (phi - om)), -s);
            float2 u10 = cscale(cexpi(-0.5f * (phi - om)),  s);
            float2 u11 = cscale(cexpi( 0.5f * (phi + om)),  c);
            apply1q(st, col, grp, 8 - q, u00, u01, u10, u11);
            __syncthreads();
        }
        // ring of CNOTs, range r = (l % 8) + 1
        const int r = (l % (NQ - 1)) + 1;
        for (int q = 0; q < NQ; ++q) {
            applyCnot(st, col, grp, 8 - q, 8 - ((q + r) % NQ));
            __syncthreads();
        }
    }

    // write f16 real/imag planes, row-major [row j][col k]
    for (int j = grp; j < DIM; j += 8) {
        Wr[j * DIM + colBase + col] = (_Float16)st[j][col].x;
        Wi[j * DIM + colBase + col] = (_Float16)st[j][col].y;
    }
}

// ===========================================================================
// Kernel 2: per 64-patch block — build Psi tile (closed form), complex GEMM
// Phi = W * Psi with v_wmma_f32_16x16x32_f16, reduce |Phi|^2 with Z0 sign.
// ===========================================================================

__device__ __forceinline__ v16h cat16(v8h a, v8h b) {
    return __builtin_shufflevector(a, b, 0, 1, 2, 3, 4, 5, 6, 7,
                                         8, 9, 10, 11, 12, 13, 14, 15);
}

__global__ __launch_bounds__(256) void qconv_wmma(const float* __restrict__ x,
                                                  const _Float16* __restrict__ Wr,
                                                  const _Float16* __restrict__ Wi,
                                                  float* __restrict__ out) {
    // 4 patch-tiles of 16 patches; Psi stored [patch][K] so B fragments are
    // contiguous (K runs fastest) -> b128 ds loads.
    __shared__ _Float16 sPsiR[4][16][DIM];         // 64 KB
    __shared__ _Float16 sPsiI[4][16][DIM];         // 64 KB
    __shared__ float sC[64][NQ], sS[64][NQ];       // cos/sin(theta/2)
    __shared__ float zacc[64];

    const int tid = threadIdx.x;
    const int patchBase = blockIdx.x * 64;

    // ---- phase 1: gather angles (3x3 unfold, pad=1) -> cos/sin tables ----
    for (int t = tid; t < 64 * NQ; t += 256) {
        const int p = t / NQ, q = t - p * NQ;
        const int flat = patchBase + p;
        const int ww = flat & (HDIM - 1);
        const int hh = (flat >> 6) & (HDIM - 1);
        const int b  = flat >> 12;
        const int kh = q / 3, kw = q - kh * 3;
        const int hi = hh + kh - 1, wi = ww + kw - 1;
        float ang = 0.f;
        if (hi >= 0 && hi < HDIM && wi >= 0 && wi < HDIM)
            ang = x[(b * HDIM + hi) * HDIM + wi];
        sC[p][q] = __cosf(0.5f * ang);
        sS[p][q] = __sinf(0.5f * ang);
    }
    if (tid < 64) zacc[tid] = 0.f;
    __syncthreads();

    // ---- phase 2: Psi_j = (prod c/s) * (-i)^popcount(j) ----
    for (int t = tid; t < 64 * DIM; t += 256) {
        const int p = t >> 9, j = t & (DIM - 1);
        float m = 1.f;
        #pragma unroll
        for (int q = 0; q < NQ; ++q)
            m *= ((j >> (8 - q)) & 1) ? sS[p][q] : sC[p][q];
        const int n = __popc(j) & 3;                    // (-i)^n
        const float pr = (n == 0) ? m : ((n == 2) ? -m : 0.f);
        const float pi = (n == 3) ? m : ((n == 1) ? -m : 0.f);
        sPsiR[p >> 4][p & 15][j] = (_Float16)pr;
        sPsiI[p >> 4][p & 15][j] = (_Float16)pi;
    }
    __syncthreads();

    // ---- phase 3: WMMA GEMM.
    // Each wave: 4 row-tiles (rt = wave, wave+8, wave+16, wave+24), all 4
    // patch tiles. One A-fragment pair (Wr/Wi) feeds 16 WMMAs.
    const int wave = tid >> 5, lane = tid & 31;
    const int m16  = lane & 15, half = lane >> 4;

    for (int rt = wave; rt < 32; rt += 8) {
        const int row = rt * 16 + m16;
        // A fragment (16x32 f16): lane m in 0..15 = row m, K {0..7,16..23};
        // lanes 16..31 = same rows, K {8..15,24..31}.
        const _Float16* arBase = Wr + row * DIM + half * 8;
        const _Float16* aiBase = Wi + row * DIM + half * 8;

        v8f accR[4] = {}, accI[4] = {};
        for (int kc = 0; kc < 16; ++kc) {              // K = 512 in chunks of 32
            const int k0 = kc * 32;
            // prefetch next k-chunk of both W planes (global_prefetch_b8)
            if (kc < 15) {
                __builtin_prefetch(arBase + k0 + 32, 0, 3);
                __builtin_prefetch(aiBase + k0 + 32, 0, 3);
            }
            v16h ar = cat16(*(const v8h*)(arBase + k0),
                            *(const v8h*)(arBase + k0 + 16));
            v16h ai = cat16(*(const v8h*)(aiBase + k0),
                            *(const v8h*)(aiBase + k0 + 16));
            v16h ain = -ai;                            // f16 WMMA NEG only covers C

            const int koff = k0 + half * 8;
            #pragma unroll
            for (int pt = 0; pt < 4; ++pt) {
                // B fragment (32x16 f16): lane col = lane%16, K half*8+{0..7,16..23}
                const _Float16* brP = &sPsiR[pt][m16][koff];
                const _Float16* biP = &sPsiI[pt][m16][koff];
                v16h br = cat16(*(const v8h*)brP, *(const v8h*)(brP + 16));
                v16h bi = cat16(*(const v8h*)biP, *(const v8h*)(biP + 16));

                // Phi_r += Wr*Pr - Wi*Pi ; Phi_i += Wr*Pi + Wi*Pr
                accR[pt] = __builtin_amdgcn_wmma_f32_16x16x32_f16(
                               false, ar,  false, br, (short)0, accR[pt],
                               false, false);
                accR[pt] = __builtin_amdgcn_wmma_f32_16x16x32_f16(
                               false, ain, false, bi, (short)0, accR[pt],
                               false, false);
                accI[pt] = __builtin_amdgcn_wmma_f32_16x16x32_f16(
                               false, ar,  false, bi, (short)0, accI[pt],
                               false, false);
                accI[pt] = __builtin_amdgcn_wmma_f32_16x16x32_f16(
                               false, ai,  false, br, (short)0, accI[pt],
                               false, false);
            }
        }

        // <Z0>: sign by MSB of row index; uniform within a 16-row tile.
        const float sgn = (rt < 16) ? 1.f : -1.f;
        #pragma unroll
        for (int pt = 0; pt < 4; ++pt) {
            float partial = 0.f;
            #pragma unroll
            for (int r2 = 0; r2 < 8; ++r2)
                partial += accR[pt][r2] * accR[pt][r2]
                         + accI[pt][r2] * accI[pt][r2];
            atomicAdd(&zacc[pt * 16 + m16], sgn * partial);
        }
    }
    __syncthreads();

    if (tid < 64) out[patchBase + tid] = zacc[tid];
}

// ===========================================================================
// launch
// ===========================================================================
extern "C" void kernel_launch(void* const* d_in, const int* in_sizes, int n_in,
                              void* d_out, int out_size, void* d_ws, size_t ws_size,
                              hipStream_t stream) {
    const float* x = (const float*)d_in[0];   // (8,1,64,64) f32
    const float* w = (const float*)d_in[1];   // (2,9,3) f32
    float* out = (float*)d_out;               // (8,1,64,64) f32

    _Float16* Wr = (_Float16*)d_ws;           // 512 KB
    _Float16* Wi = Wr + DIM * DIM;            // 512 KB  (ws >= 1 MB)

    build_W<<<DIM / 32, 256, 0, stream>>>(w, Wr, Wi);

    const int nBlocks = out_size / 64;        // 32768/64 = 512
    qconv_wmma<<<nBlocks, 256, 0, stream>>>(x, Wr, Wi, out);
}
